// GCN_84310208020845
// MI455X (gfx1250) — matrix operand order repaired and checked by reading, not compiled, not run on hardware
//
#include <hip/hip_runtime.h>
#include <hip/hip_bf16.h>

// ---------------------------------------------------------------------------
// Problem constants (from reference)
// ---------------------------------------------------------------------------
#define BB   4
#define TT   220
#define CC   512
#define NHH  256
#define NCC  768
#define NUMM 1296
#define BT   (BB * TT)          // 880
#define MAPO 2048

typedef __attribute__((ext_vector_type(16))) _Float16 v16h;
typedef __attribute__((ext_vector_type(4)))  _Float16 v4h;
typedef __attribute__((ext_vector_type(8)))  float    v8f;
typedef __attribute__((ext_vector_type(4)))  float    v4f;

// ---------------------------------------------------------------------------
// Generic GEMM: C[M,N] = A[M,K] @ B[K,N] (+bias) via v_wmma_f32_16x16x32_f16.
// f32 in memory, f16 compute, f32 accumulate. K must be a multiple of 32.
//
// LDS tiles are stored in *fragment order* so each lane's WMMA operand is a
// contiguous 32-byte run (two ds_load_b128):
//   A (16-bit 16x32, ISA 7.12.2): lane group g = lane>>4 needs
//       k = h + 16g for h = 0..15 (contiguous, same order as B)
//   -> store k at pos(k) = k (no permutation); fragment = [16g..16g+15]
//   B (32x16): lane group g uses k = h + 16g, same order as A
//   -> store transposed Bs[col][k]; fragment = [16g..16g+15]
//
// Out-of-range rows/cols are CLAMPED (not zeroed): clamped data only feeds
// output rows/cols that the epilogue masks, so staging loads are unconditional.
// Staging is split into a load phase (issued before the barrier, overlapping
// the previous iteration's WMMAs) and a convert/store phase. Global addresses
// are carried as size_t offsets from the kernel-arg base pointers so the
// compiler keeps global addressing (global_load_*, not flat_load_* which
// would tick DScnt and contend with the LDS pipe).
// permuteTB: remap output row r=(b*T+t) -> t*B+b (final transpose fusion).
// Block: 128 threads (4 waves), 64x64 tile; each wave owns 32x32 (2x2 WMMA).
// ---------------------------------------------------------------------------
__global__ __launch_bounds__(128) void gemm_wmma_f16(
    const float* __restrict__ A, const float* __restrict__ Bw,
    float* __restrict__ C, const float* __restrict__ bias,
    int M, int N, int K, int permuteTB)
{
    __shared__ _Float16 As[64][40];   // [row][pos(k)], pitch 80B (16B aligned)
    __shared__ _Float16 Bs[64][40];   // [col][k]

    const int tid  = threadIdx.x;
    const int wave = tid >> 5;
    const int lane = tid & 31;
    const int g    = lane >> 4;      // half-wave group
    const int mi   = lane & 15;
    const int wm   = (wave >> 1) * 32;
    const int wn   = (wave & 1) * 32;

    const int bm = blockIdx.y * 64;
    const int bn = blockIdx.x * 64;

    const v8f zero8 = {0.f, 0.f, 0.f, 0.f, 0.f, 0.f, 0.f, 0.f};
    v8f acc[2][2];
    acc[0][0] = zero8; acc[0][1] = zero8; acc[1][0] = zero8; acc[1][1] = zero8;

    // Per-thread staging offsets (advanced by 32 / 32*N per K-step).
    // A: idx -> row = idx>>3, c4 = (idx&7)*4   (512 float4 loads, 4/thread)
    // B: idx -> col = idx&63, k4 = (idx>>6)*4  (coalesced column loads)
    size_t aoff[4]; v4h* alp[4];
    size_t boff[4]; v4h* blp[4];
    #pragma unroll
    for (int q = 0; q < 4; ++q) {
        int idx = q * 128 + tid;
        int r   = idx >> 3;
        int c4  = (idx & 7) * 4;
        int gr  = bm + r; if (gr > M - 1) gr = M - 1;
        aoff[q] = (size_t)gr * K + c4;
        int pos = c4;
        alp[q] = (v4h*)&As[r][pos];
        int col = idx & 63;
        int k4  = (idx >> 6) * 4;
        int gc  = bn + col; if (gc > N - 1) gc = N - 1;
        boff[q] = (size_t)k4 * N + gc;
        blp[q] = (v4h*)&Bs[col][k4];
    }

    for (int k0 = 0; k0 < K; k0 += 32) {
        // ---- load phase: all global reads issued before the barrier ----
        v4f  av[4];
        float bv[4][4];
        #pragma unroll
        for (int q = 0; q < 4; ++q)
            av[q] = *(const v4f*)(A + aoff[q]);
        #pragma unroll
        for (int q = 0; q < 4; ++q)
            #pragma unroll
            for (int kk = 0; kk < 4; ++kk)
                bv[q][kk] = Bw[boff[q] + (size_t)kk * N];
        #pragma unroll
        for (int q = 0; q < 4; ++q) {
            aoff[q] += 32;
            boff[q] += (size_t)32 * N;
        }
        // Prefetch next K tile (global_prefetch_b8)
        if (k0 + 32 < K) {
            __builtin_prefetch(A + aoff[0], 0, 1);
            __builtin_prefetch(Bw + boff[0], 0, 1);
        }
        __syncthreads();   // previous iteration's fragment reads complete
        // ---- convert + store phase ----
        #pragma unroll
        for (int q = 0; q < 4; ++q) {
            v4h h4;
            h4[0] = (_Float16)av[q][0]; h4[1] = (_Float16)av[q][1];
            h4[2] = (_Float16)av[q][2]; h4[3] = (_Float16)av[q][3];
            *alp[q] = h4;
        }
        #pragma unroll
        for (int q = 0; q < 4; ++q) {
            v4h h4;
            h4[0] = (_Float16)bv[q][0]; h4[1] = (_Float16)bv[q][1];
            h4[2] = (_Float16)bv[q][2]; h4[3] = (_Float16)bv[q][3];
            *blp[q] = h4;
        }
        __syncthreads();
        // ---- fragments + WMMA ----
        v16h af[2], bf[2];
        #pragma unroll
        for (int tm = 0; tm < 2; ++tm)
            af[tm] = *(const v16h*)&As[wm + tm * 16 + mi][16 * g];
        #pragma unroll
        for (int tn = 0; tn < 2; ++tn)
            bf[tn] = *(const v16h*)&Bs[wn + tn * 16 + mi][16 * g];

        #pragma unroll
        for (int tm = 0; tm < 2; ++tm)
            #pragma unroll
            for (int tn = 0; tn < 2; ++tn)
                acc[tm][tn] = __builtin_amdgcn_wmma_f32_16x16x32_f16(
                    false, af[tm], false, bf[tn], (short)0, acc[tm][tn], false, false);
    }

    // Epilogue: C layout is M = r + 8*g, N = lane&15 per 16x16 tile
    #pragma unroll
    for (int tm = 0; tm < 2; ++tm)
        #pragma unroll
        for (int tn = 0; tn < 2; ++tn)
            #pragma unroll
            for (int r = 0; r < 8; ++r) {
                int row = bm + wm + tm * 16 + r + g * 8;
                int col = bn + wn + tn * 16 + mi;
                if (row < M && col < N) {
                    float v = acc[tm][tn][r];
                    if (bias) v += bias[col];
                    int orow = row;
                    if (permuteTB) orow = (row % TT) * BB + (row / TT);
                    C[(size_t)orow * N + col] = v;
                }
            }
}

// ---------------------------------------------------------------------------
// f1[r] = H[r,:]·a[0:D], f2[r] = H[r,:]·a[D:2D]   (one block per row)
// ---------------------------------------------------------------------------
__global__ __launch_bounds__(256) void gat_fvec(
    const float* __restrict__ H, const float* __restrict__ a,
    float* __restrict__ f1, float* __restrict__ f2, int D)
{
    int row = blockIdx.x, tid = threadIdx.x;
    __shared__ float s1[256], s2[256];
    float p1 = 0.f, p2 = 0.f;
    for (int c = tid; c < D; c += 256) {
        float h = H[(size_t)row * D + c];
        p1 += h * a[c];
        p2 += h * a[D + c];
    }
    s1[tid] = p1; s2[tid] = p2;
    __syncthreads();
    for (int s = 128; s > 0; s >>= 1) {
        if (tid < s) { s1[tid] += s1[tid + s]; s2[tid] += s2[tid + s]; }
        __syncthreads();
    }
    if (tid == 0) { f1[row] = s1[0]; f2[row] = s2[0]; }
}

// ---------------------------------------------------------------------------
// Banded (|i-j|<=2, j!=i) attention + ELU. One block per (b,i) row.
// ---------------------------------------------------------------------------
__global__ __launch_bounds__(256) void band_attn_elu(
    const float* __restrict__ H, const float* __restrict__ f1,
    const float* __restrict__ f2, float* __restrict__ O, int D)
{
    int row = blockIdx.x;
    int b = row / TT, i = row % TT, base = b * TT;
    float w[4]; int js[4]; int nn = 0;
    #pragma unroll
    for (int dj = -2; dj <= 2; ++dj) {
        if (dj == 0) continue;
        int j = i + dj;
        if (j < 0 || j >= TT) continue;
        js[nn] = base + j;
        float e = f1[row] + f2[base + j];
        w[nn] = e > 0.f ? e : 0.01f * e;   // leaky_relu
        ++nn;
    }
    float mx = -1e30f;
    for (int k = 0; k < nn; ++k) mx = fmaxf(mx, w[k]);
    float sm = 0.f;
    for (int k = 0; k < nn; ++k) { w[k] = __expf(w[k] - mx); sm += w[k]; }
    float inv = 1.f / sm;
    for (int c = threadIdx.x; c < D; c += 256) {
        float o = 0.f;
        for (int k = 0; k < nn; ++k) o += w[k] * H[(size_t)js[k] * D + c];
        o *= inv;
        O[(size_t)row * D + c] = o > 0.f ? o : (__expf(o) - 1.f);  // elu
    }
}

// ---------------------------------------------------------------------------
// Banded attention + ELU + log_softmax over D=768. One block (256 thr) per row.
// ---------------------------------------------------------------------------
__global__ __launch_bounds__(256) void band_attn_elu_lsm(
    const float* __restrict__ H, const float* __restrict__ f1,
    const float* __restrict__ f2, float* __restrict__ O)
{
    const int D = NCC;
    int row = blockIdx.x, tid = threadIdx.x;
    int b = row / TT, i = row % TT, base = b * TT;
    float w[4]; int js[4]; int nn = 0;
    #pragma unroll
    for (int dj = -2; dj <= 2; ++dj) {
        if (dj == 0) continue;
        int j = i + dj;
        if (j < 0 || j >= TT) continue;
        js[nn] = base + j;
        float e = f1[row] + f2[base + j];
        w[nn] = e > 0.f ? e : 0.01f * e;
        ++nn;
    }
    float mx = -1e30f;
    for (int k = 0; k < nn; ++k) mx = fmaxf(mx, w[k]);
    float sm = 0.f;
    for (int k = 0; k < nn; ++k) { w[k] = __expf(w[k] - mx); sm += w[k]; }
    float inv = 1.f / sm;

    float y[3];
    #pragma unroll
    for (int q = 0; q < 3; ++q) {
        int c = tid + q * 256;
        float o = 0.f;
        for (int k = 0; k < nn; ++k) o += w[k] * H[(size_t)js[k] * D + c];
        o *= inv;
        y[q] = o > 0.f ? o : (__expf(o) - 1.f);
    }
    __shared__ float red[256];
    float lm = fmaxf(fmaxf(y[0], y[1]), y[2]);
    red[tid] = lm; __syncthreads();
    for (int s = 128; s > 0; s >>= 1) {
        if (tid < s) red[tid] = fmaxf(red[tid], red[tid + s]);
        __syncthreads();
    }
    float gm = red[0]; __syncthreads();
    float ps = __expf(y[0] - gm) + __expf(y[1] - gm) + __expf(y[2] - gm);
    red[tid] = ps; __syncthreads();
    for (int s = 128; s > 0; s >>= 1) {
        if (tid < s) red[tid] += red[tid + s];
        __syncthreads();
    }
    float lse = gm + __logf(red[0]);
    #pragma unroll
    for (int q = 0; q < 3; ++q)
        O[(size_t)row * D + tid + q * 256] = y[q] - lse;
}

// ---------------------------------------------------------------------------
// Keypoint gather from pose_map + zero-mask + L2-normalize 4 node vectors.
// One block (256 thr) per bt; each thread owns channels tid and tid+256.
// ---------------------------------------------------------------------------
__global__ __launch_bounds__(256) void gather_norm(
    const float* __restrict__ pm, const float* __restrict__ pv,
    const float* __restrict__ fkp, const float* __restrict__ lkp,
    const float* __restrict__ rkp, float* __restrict__ nodes)
{
    int bt = blockIdx.x, tid = threadIdx.x;
    auto ki = [](float v) {
        int x = (int)(v * (7.0f / 256.0f)) - 1;
        return x < 0 ? 0 : (x > 6 ? 6 : x);
    };
    int fr = ki(fkp[bt * 2]), fc = ki(fkp[bt * 2 + 1]);
    int lr = ki(lkp[bt * 2]), lc = ki(lkp[bt * 2 + 1]);
    int rr = ki(rkp[bt * 2]), rc = ki(rkp[bt * 2 + 1]);
    bool lz = (lkp[bt * 2] == 0.f) && (lkp[bt * 2 + 1] == 0.f);
    bool rz = (rkp[bt * 2] == 0.f) && (rkp[bt * 2 + 1] == 0.f);

    size_t pmb = (size_t)bt * CC * 49;
    int c0 = tid, c1 = tid + 256;
    float p0 = pv[(size_t)bt * CC + c0], p1 = pv[(size_t)bt * CC + c1];
    float fa0 = pm[pmb + (size_t)c0 * 49 + fr * 7 + fc];
    float fa1 = pm[pmb + (size_t)c1 * 49 + fr * 7 + fc];
    float l0 = lz ? 0.f : pm[pmb + (size_t)c0 * 49 + lr * 7 + lc];
    float l1 = lz ? 0.f : pm[pmb + (size_t)c1 * 49 + lr * 7 + lc];
    float r0 = rz ? 0.f : pm[pmb + (size_t)c0 * 49 + rr * 7 + rc];
    float r1 = rz ? 0.f : pm[pmb + (size_t)c1 * 49 + rr * 7 + rc];

    __shared__ float red[256];
    __shared__ float bsum[4];
    float ss[4] = {p0 * p0 + p1 * p1, fa0 * fa0 + fa1 * fa1,
                   r0 * r0 + r1 * r1, l0 * l0 + l1 * l1};
    #pragma unroll
    for (int p = 0; p < 4; ++p) {
        red[tid] = ss[p]; __syncthreads();
        for (int s = 128; s > 0; s >>= 1) {
            if (tid < s) red[tid] += red[tid + s];
            __syncthreads();
        }
        if (tid == 0) bsum[p] = red[0];
        __syncthreads();
    }
    float ip  = 1.f / (sqrtf(bsum[0]) + 1e-12f);
    float ifa = 1.f / (sqrtf(bsum[1]) + 1e-12f);
    float ir  = 1.f / (sqrtf(bsum[2]) + 1e-12f);
    float il  = 1.f / (sqrtf(bsum[3]) + 1e-12f);
    size_t nb = (size_t)bt * 4 * CC;
    nodes[nb + c0]          = p0 * ip;  nodes[nb + c1]          = p1 * ip;
    nodes[nb + CC + c0]     = fa0 * ifa; nodes[nb + CC + c1]     = fa1 * ifa;
    nodes[nb + 2 * CC + c0] = r0 * ir;  nodes[nb + 2 * CC + c1] = r1 * ir;
    nodes[nb + 3 * CC + c0] = l0 * il;  nodes[nb + 3 * CC + c1] = l1 * il;
}

// ---------------------------------------------------------------------------
// Euclidean 4x4 attention per frame: softmax(lrelu(f1+f2^T) - a0*dist) @ h, ELU
// ---------------------------------------------------------------------------
__global__ __launch_bounds__(256) void euli_attn(
    const float* __restrict__ Hm, const float* __restrict__ fm1,
    const float* __restrict__ fm2, const float* __restrict__ alpha0,
    float* __restrict__ Gm)
{
    const int D = NCC;
    int bt = blockIdx.x, tid = threadIdx.x;
    const float* h = Hm + (size_t)bt * 4 * D;

    float d2[6] = {0.f, 0.f, 0.f, 0.f, 0.f, 0.f};
    for (int c = tid; c < D; c += 256) {
        float h0 = h[c], h1 = h[D + c], h2 = h[2 * D + c], h3 = h[3 * D + c], t;
        t = h0 - h1; d2[0] += t * t;
        t = h0 - h2; d2[1] += t * t;
        t = h0 - h3; d2[2] += t * t;
        t = h1 - h2; d2[3] += t * t;
        t = h1 - h3; d2[4] += t * t;
        t = h2 - h3; d2[5] += t * t;
    }
    __shared__ float red[256];
    __shared__ float dsh[6];
    __shared__ float attn[16];
    #pragma unroll
    for (int p = 0; p < 6; ++p) {
        red[tid] = d2[p]; __syncthreads();
        for (int s = 128; s > 0; s >>= 1) {
            if (tid < s) red[tid] += red[tid + s];
            __syncthreads();
        }
        if (tid == 0) dsh[p] = red[0];
        __syncthreads();
    }
    if (tid == 0) {
        float a0 = alpha0[0];
        float dd[6];
        for (int p = 0; p < 6; ++p) dd[p] = sqrtf(dsh[p] + 1e-12f);
        float dist[4][4];
        float diag = sqrtf(1e-12f);
        dist[0][0] = dist[1][1] = dist[2][2] = dist[3][3] = diag;
        dist[0][1] = dist[1][0] = dd[0];
        dist[0][2] = dist[2][0] = dd[1];
        dist[0][3] = dist[3][0] = dd[2];
        dist[1][2] = dist[2][1] = dd[3];
        dist[1][3] = dist[3][1] = dd[4];
        dist[2][3] = dist[3][2] = dd[5];
        for (int i2 = 0; i2 < 4; ++i2) {
            float rw[4], mx = -1e30f;
            for (int j = 0; j < 4; ++j) {
                float e = fm1[bt * 4 + i2] + fm2[bt * 4 + j];
                e = e > 0.f ? e : 0.01f * e;
                rw[j] = e - a0 * dist[i2][j];
                mx = fmaxf(mx, rw[j]);
            }
            float sm = 0.f;
            for (int j = 0; j < 4; ++j) { rw[j] = __expf(rw[j] - mx); sm += rw[j]; }
            for (int j = 0; j < 4; ++j) attn[i2 * 4 + j] = rw[j] / sm;
        }
    }
    __syncthreads();
    for (int c = tid; c < D; c += 256) {
        float h0 = h[c], h1 = h[D + c], h2 = h[2 * D + c], h3 = h[3 * D + c];
        #pragma unroll
        for (int i2 = 0; i2 < 4; ++i2) {
            float o = attn[i2 * 4] * h0 + attn[i2 * 4 + 1] * h1 +
                      attn[i2 * 4 + 2] * h2 + attn[i2 * 4 + 3] * h3;
            Gm[(size_t)bt * (4 * NCC) + i2 * NCC + c] = o > 0.f ? o : (__expf(o) - 1.f);
        }
    }
}

// ---------------------------------------------------------------------------
// final[r, 0:2048] = fused[r], final[r, 2048:2816] = ff[r]
// ---------------------------------------------------------------------------
__global__ __launch_bounds__(256) void concat_final(
    const float* __restrict__ fused, const float* __restrict__ ff,
    float* __restrict__ fin)
{
    size_t idx = (size_t)blockIdx.x * 256 + threadIdx.x;
    const size_t total = (size_t)BT * (MAPO + NCC);
    if (idx >= total) return;
    size_t r = idx / (MAPO + NCC);
    int c = (int)(idx % (MAPO + NCC));
    fin[idx] = (c < MAPO) ? fused[r * MAPO + c] : ff[r * NCC + (c - MAPO)];
}

// ---------------------------------------------------------------------------
// Host-side launch
// ---------------------------------------------------------------------------
extern "C" void kernel_launch(void* const* d_in, const int* in_sizes, int n_in,
                              void* d_out, int out_size, void* d_ws, size_t ws_size,
                              hipStream_t stream)
{
    const float* pose_map = (const float*)d_in[0];
    const float* pose_vec = (const float*)d_in[1];
    const float* face_kp  = (const float*)d_in[2];
    const float* lhand_kp = (const float*)d_in[3];
    const float* rhand_kp = (const float*)d_in[4];
    const float* W1       = (const float*)d_in[5];
    const float* a1       = (const float*)d_in[6];
    const float* W2       = (const float*)d_in[7];
    const float* a2       = (const float*)d_in[8];
    const float* Wm       = (const float*)d_in[9];
    const float* am       = (const float*)d_in[10];
    const float* alpha0   = (const float*)d_in[11];
    const float* map_w    = (const float*)d_in[12];
    const float* map_b    = (const float*)d_in[13];
    const float* cls_w    = (const float*)d_in[14];
    const float* cls_b    = (const float*)d_in[15];
    float* out = (float*)d_out;

    float* ws = (float*)d_ws;
    size_t o = 0;
    float* h1    = ws + o; o += (size_t)BT * NHH;        // 880x256
    float* f11   = ws + o; o += BT;
    float* f21   = ws + o; o += BT;
    float* g1    = ws + o; o += (size_t)BT * NHH;        // 880x256
    float* h2    = ws + o; o += (size_t)BT * NCC;        // 880x768
    float* f12   = ws + o; o += BT;
    float* f22   = ws + o; o += BT;
    float* ff    = ws + o; o += (size_t)BT * NCC;        // 880x768
    float* nodes = ws + o; o += (size_t)BT * 4 * CC;     // 3520x512
    float* Hm    = ws + o; o += (size_t)BT * 4 * NCC;    // 3520x768
    float* fm1   = ws + o; o += (size_t)BT * 4;
    float* fm2   = ws + o; o += (size_t)BT * 4;
    float* Gm    = ws + o; o += (size_t)BT * 4 * NCC;    // 880x3072
    float* fused = ws + o; o += (size_t)BT * MAPO;       // 880x2048
    float* finb  = ws + o; o += (size_t)BT * (MAPO + NCC); // 880x2816
    (void)ws_size; (void)in_sizes; (void)n_in; (void)out_size;

    auto gemm = [&](const float* A, const float* Bw, float* Cout, const float* bias,
                    int M, int N, int K, int permute) {
        dim3 grid((N + 63) / 64, (M + 63) / 64);
        gemm_wmma_f16<<<grid, 128, 0, stream>>>(A, Bw, Cout, bias, M, N, K, permute);
    };

    // --- spatial GAT layer 1 ---
    gemm(pose_vec, W1, h1, nullptr, BT, NHH, CC, 0);
    gat_fvec<<<BT, 256, 0, stream>>>(h1, a1, f11, f21, NHH);
    band_attn_elu<<<BT, 256, 0, stream>>>(h1, f11, f21, g1, NHH);
    // --- spatial GAT layer 2 + log_softmax ---
    gemm(g1, W2, h2, nullptr, BT, NCC, NHH, 0);
    gat_fvec<<<BT, 256, 0, stream>>>(h2, a2, f12, f22, NCC);
    band_attn_elu_lsm<<<BT, 256, 0, stream>>>(h2, f12, f22, ff);
    // --- keypoint gather + normalize ---
    gather_norm<<<BT, 256, 0, stream>>>(pose_map, pose_vec, face_kp, lhand_kp,
                                        rhand_kp, nodes);
    // --- Euclidean GAT ---
    gemm(nodes, Wm, Hm, nullptr, BT * 4, NCC, CC, 0);
    gat_fvec<<<BT * 4, 256, 0, stream>>>(Hm, am, fm1, fm2, NCC);
    euli_attn<<<BT, 256, 0, stream>>>(Hm, fm1, fm2, alpha0, Gm);
    // --- fusion + classifier (with (B,T)->(T,B) permute fused into store) ---
    gemm(Gm, map_w, fused, map_b, BT, MAPO, 4 * NCC, 0);
    size_t tot = (size_t)BT * (MAPO + NCC);
    concat_final<<<(unsigned)((tot + 255) / 256), 256, 0, stream>>>(fused, ff, finb);
    gemm(finb, cls_w, out, cls_b, BT, NUMM, MAPO + NCC, 1);
}
